// Elman_28295244546270
// MI455X (gfx1250) — matrix operand order repaired and checked by reading, not compiled, not run on hardware
//
#include <hip/hip_runtime.h>
#include <hip/hip_bf16.h>
#include <math.h>
#include <stdint.h>

#define B_ 64
#define T_ 512
#define H_ 1024
#define NWG_REC 16

typedef __bf16 v16bf __attribute__((ext_vector_type(16)));
typedef float  v8f   __attribute__((ext_vector_type(8)));
typedef unsigned int u32x4 __attribute__((ext_vector_type(4)));
typedef int i32x4 __attribute__((ext_vector_type(4)));
typedef int i32x8 __attribute__((ext_vector_type(8)));

union FragBF { v16bf v; uint4 u[2]; };

// A-matrix 16x32 bf16 fragment from a K-major (row-major [rows, K]) matrix.
// lane<16: row r, K = {k0..k0+7, k0+16..k0+23}; lane>=16: K = {+8..+15, +24..+31}.
__device__ __forceinline__ v16bf load_a_frag(const __bf16* __restrict__ base, int ld,
                                             int row0, int k0, int lane) {
  const int r = lane & 15, hi = lane >> 4;
  const __bf16* q = base + (size_t)(row0 + r) * ld + k0 + (hi << 3);
  FragBF f;
  f.u[0] = *(const uint4*)(q);
  f.u[1] = *(const uint4*)(q + 16);
  return f.v;
}

// B-matrix 32x16 bf16 fragment: lane<16 = column c, K = {k0..k0+7, k0+16..k0+23};
// lane>=16 = column c, K = {+8..+15, +24..+31}. Source stored [N, K] row-major.
__device__ __forceinline__ v16bf load_b_frag(const __bf16* __restrict__ base, int ld,
                                             int col0, int k0, int lane) {
  const int r = lane & 15, hi = lane >> 4;
  const __bf16* q = base + (size_t)(col0 + r) * ld + k0 + (hi << 3);
  FragBF f;
  f.u[0] = *(const uint4*)(q);
  f.u[1] = *(const uint4*)(q + 16);
  return f.v;
}

__global__ __launch_bounds__(256) void cvt_bf16_kernel(const float* __restrict__ s,
                                                       __bf16* __restrict__ d, int n) {
  int i = blockIdx.x * blockDim.x + threadIdx.x;
  int st = gridDim.x * blockDim.x;
  for (; i < n; i += st) d[i] = (__bf16)s[i];
}

// Phase 1: out[m, n] = sum_k Xb[m, k] * W1b[n, k] + b1[n]
// Block = 256 threads = 8 waves; WG tile 128(M) x 128(N); wave tile 64(M) x 32(N).
// All 6 fragment loads of a K-step are hoisted ahead of the 8 WMMAs so loads
// overlap compute and operand regs aren't recycled into the WAR-hazard window.
__global__ __launch_bounds__(256) void gemm1_kernel(const __bf16* __restrict__ Xb,
                                                    const __bf16* __restrict__ W1b,
                                                    const float* __restrict__ b1,
                                                    float* __restrict__ out) {
  const int lane = threadIdx.x & 31;
  const int wave = threadIdx.x >> 5;
  const int wgn = blockIdx.x & 7;    // 8 N-blocks of 128
  const int wgm = blockIdx.x >> 3;   // 256 M-blocks of 128
  const int mq = wave & 1;
  const int nq = wave >> 1;          // 0..3
  const int m0 = wgm * 128 + mq * 64;
  const int n0 = wgn * 128 + nq * 32;
  const int r = lane & 15, hi = lane >> 4;

  v8f c[4][2] = {};
  for (int k0 = 0; k0 < H_; k0 += 32) {
    v16bf bf0 = load_b_frag(W1b, H_, n0, k0, lane);
    v16bf bf1 = load_b_frag(W1b, H_, n0 + 16, k0, lane);
    v16bf af[4];
#pragma unroll
    for (int mt = 0; mt < 4; ++mt) {
      af[mt] = load_a_frag(Xb, H_, m0 + mt * 16, k0, lane);
    }
#pragma unroll
    for (int mt = 0; mt < 4; ++mt) {
      c[mt][0] = __builtin_amdgcn_wmma_f32_16x16x32_bf16(false, af[mt], false, bf0,
                                                         (short)0, c[mt][0], false, false);
      c[mt][1] = __builtin_amdgcn_wmma_f32_16x16x32_bf16(false, af[mt], false, bf1,
                                                         (short)0, c[mt][1], false, false);
    }
  }
#pragma unroll
  for (int nt = 0; nt < 2; ++nt) {
    const int nn = n0 + nt * 16 + r;
    const float bias = b1[nn];
#pragma unroll
    for (int mt = 0; mt < 4; ++mt) {
#pragma unroll
      for (int v = 0; v < 8; ++v) {
        int m = m0 + mt * 16 + v + (hi << 3);
        out[(size_t)m * H_ + nn] = c[mt][nt][v] + bias;
      }
    }
  }
}

// Device-wide barrier: HW cluster barrier (S_NOP when not launched as a
// cluster) + monotonic atomic counter for guaranteed correctness.
__device__ __forceinline__ void grid_bar(unsigned int* bar, unsigned int target) {
  __syncthreads();
  if (threadIdx.x < 32) {
    __builtin_amdgcn_s_cluster_barrier();  // s_barrier_signal -3 ; s_barrier_wait -3
  }
  if (threadIdx.x == 0) {
    atomicAdd(bar, 1u);
    while (__hip_atomic_load(bar, __ATOMIC_RELAXED, __HIP_MEMORY_SCOPE_AGENT) < target) {
      __builtin_amdgcn_s_sleep(2);
    }
  }
  __syncthreads();
}

// TDM: DMA h[:, col0:col0+512] (64 x 512 bf16, row stride 1024) into LDS.
// 2D descriptor per CDNA5 ISA ch.8: group0 = {flags, lds_addr, global_addr, type},
// group1 = {mask/data_size, dims, tile dims, strides}. Groups 2/3 unused (zeros).
// 6-arg builtin on this toolchain: (u32x4, i32x8, i32x4, i32x4, i32x8, i32 cpol).
__device__ __forceinline__ void tdm_issue_h_half(const __bf16* gsrc, __bf16* lds_dst) {
  unsigned int lds_lo = (unsigned int)(uintptr_t)lds_dst;
  unsigned long long ga = (unsigned long long)(uintptr_t)gsrc;
  u32x4 g0 = {1u,                                    // count=1, user descriptor
              lds_lo,                                // lds_addr (bytes)
              (unsigned int)ga,                      // global_addr[31:0]
              (unsigned int)((ga >> 32) & 0x01FFFFFFu) | (2u << 30)};  // [56:32] + type=2
  i32x8 g1 = {(int)(1u << 16),        // workgroup_mask=0, data_size=1 (2 bytes)
              (int)(1024u << 16),     // tensor_dim0 = 1024 (low 16 in [31:16])
              (int)(64u << 16),       // tensor_dim0 hi=0, tensor_dim1 = 64
              (int)(512u << 16),      // tensor_dim1 hi=0, tile_dim0 = 512
              (int)64,                // tile_dim1 = 64, tile_dim2 = 0
              (int)1024,              // tensor_dim0_stride = 1024 (low 32)
              0, 0};
  i32x4 gz4 = {0, 0, 0, 0};
  i32x8 gz8 = {0, 0, 0, 0, 0, 0, 0, 0};
  __builtin_amdgcn_tensor_load_to_lds(g0, g1, gz4, gz4, gz8, 0);
}

// Phase 2: persistent recurrent kernel. 16 WGs x 128 threads (4 waves).
// Wave 'strip' owns output columns [strip*16, strip*16+16) for all 64 batch rows.
// h staged by the Tensor Data Mover into double-buffered LDS (2 x 64KB): both
// half-tiles are issued up front; tensorcnt<=1 gates half 0 (TDM completes
// in order), tensorcnt==0 gates half 1, overlapping DMA with WMMA compute.
__global__ __launch_bounds__(128) void recur_kernel(const __bf16* __restrict__ W2b,
                                                    const float* __restrict__ b2,
                                                    float* __restrict__ out,
                                                    __bf16* __restrict__ hbuf,
                                                    unsigned int* __restrict__ bar) {
  __shared__ __bf16 hst[2][B_ * 512];  // 128 KB (gfx1250 WGP: up to 320 KB)
  const int lane = threadIdx.x & 31;
  const int wave = threadIdx.x >> 5;
  const int strip = blockIdx.x * 4 + wave;
  const int n0 = strip * 16;
  const int r = lane & 15, hi = lane >> 4;
  const int n = n0 + r;
  const float bias = b2[n];

  // t = 0: h0 == 0, so h1 = tanh(xp_0 + b2). No GEMM, no h init needed.
  {
    __bf16* hnext = hbuf + (size_t)B_ * H_;  // buffer 1 holds h after step 0
#pragma unroll
    for (int mt = 0; mt < 4; ++mt) {
#pragma unroll
      for (int v = 0; v < 8; ++v) {
        int b = mt * 16 + v + (hi << 3);
        size_t oidx = ((size_t)b * T_ + 0) * H_ + n;
        float val = tanhf(out[oidx] + bias);
        out[oidx] = val;
        hnext[b * H_ + n] = (__bf16)val;
      }
    }
  }
  __threadfence();
  grid_bar(bar, (unsigned)NWG_REC);

  for (int t = 1; t < T_; ++t) {
    const __bf16* hprev = hbuf + (size_t)(t & 1) * B_ * H_;
    __bf16* hnext = hbuf + (size_t)((t + 1) & 1) * B_ * H_;

    // Issue both TDM half-tiles immediately; they fill LDS while we prefetch
    // x_proj and compute half 0.
    if (wave == 0) {
      tdm_issue_h_half(hprev, hst[0]);
      tdm_issue_h_half(hprev + 512, hst[1]);
      __builtin_amdgcn_s_wait_tensorcnt(1);  // half 0 resident (in-order done)
    }

    // Prefetch this step's x_proj values so the loads overlap the GEMM.
    float xr[4][8];
#pragma unroll
    for (int mt = 0; mt < 4; ++mt) {
#pragma unroll
      for (int v = 0; v < 8; ++v) {
        int b = mt * 16 + v + (hi << 3);
        xr[mt][v] = out[((size_t)b * T_ + t) * H_ + n];
      }
    }

    v8f c[4] = {};
#pragma unroll
    for (int hh = 0; hh < 2; ++hh) {
      if (hh == 1 && wave == 0) {
        __builtin_amdgcn_s_wait_tensorcnt(0);  // half 1 resident
      }
      __syncthreads();
      for (int ks = 0; ks < 16; ++ks) {
        const int kk = ks * 32;
        v16bf bf = load_b_frag(W2b, H_, n0, hh * 512 + kk, lane);
        FragBF af[4];
#pragma unroll
        for (int mt = 0; mt < 4; ++mt) {
          const __bf16* q = &hst[hh][(mt * 16 + r) * 512 + kk + (hi << 3)];
          af[mt].u[0] = *(const uint4*)(q);
          af[mt].u[1] = *(const uint4*)(q + 16);
        }
#pragma unroll
        for (int mt = 0; mt < 4; ++mt) {
          c[mt] = __builtin_amdgcn_wmma_f32_16x16x32_bf16(false, af[mt].v, false, bf,
                                                          (short)0, c[mt], false, false);
        }
      }
    }

#pragma unroll
    for (int mt = 0; mt < 4; ++mt) {
#pragma unroll
      for (int v = 0; v < 8; ++v) {
        int b = mt * 16 + v + (hi << 3);
        size_t oidx = ((size_t)b * T_ + t) * H_ + n;
        float val = tanhf(c[mt][v] + xr[mt][v] + bias);
        out[oidx] = val;
        hnext[b * H_ + n] = (__bf16)val;
      }
    }
    __threadfence();
    // grid_bar's syncthreads also fence LDS reuse before next step's TDM.
    grid_bar(bar, (unsigned)(t + 1) * NWG_REC);
  }
}

extern "C" void kernel_launch(void* const* d_in, const int* in_sizes, int n_in,
                              void* d_out, int out_size, void* d_ws, size_t ws_size,
                              hipStream_t stream) {
  const float* X  = (const float*)d_in[0];
  const float* W1 = (const float*)d_in[1];
  const float* b1 = (const float*)d_in[2];
  const float* W2 = (const float*)d_in[3];
  const float* b2 = (const float*)d_in[4];
  float* out = (float*)d_out;

  char* ws = (char*)d_ws;
  const size_t xb_bytes = (size_t)B_ * T_ * H_ * 2;  // 64 MB
  const size_t w_bytes  = (size_t)H_ * H_ * 2;       // 2 MB
  const size_t h_bytes  = (size_t)2 * B_ * H_ * 2;   // 256 KB
  __bf16* Xb  = (__bf16*)(ws);
  __bf16* W1b = (__bf16*)(ws + xb_bytes);
  __bf16* W2b = (__bf16*)(ws + xb_bytes + w_bytes);
  __bf16* hb  = (__bf16*)(ws + xb_bytes + 2 * w_bytes);
  unsigned int* bar = (unsigned int*)(ws + xb_bytes + 2 * w_bytes + h_bytes);

  (void)hipMemsetAsync(bar, 0, 256, stream);

  cvt_bf16_kernel<<<8192, 256, 0, stream>>>(X, Xb, B_ * T_ * H_);
  cvt_bf16_kernel<<<1024, 256, 0, stream>>>(W1, W1b, H_ * H_);
  cvt_bf16_kernel<<<1024, 256, 0, stream>>>(W2, W2b, H_ * H_);

  // Phase 1: x_proj -> d_out (phase 2 reads it once and overwrites in place).
  gemm1_kernel<<<2048, 256, 0, stream>>>(Xb, W1b, b1, out);

  // Phase 2: persistent recurrent kernel with per-step device-wide barrier.
  recur_kernel<<<NWG_REC, 128, 0, stream>>>(W2b, b2, out, hb, bar);
}
